// Qwen3DecoderLayer_601295421921
// MI455X (gfx1250) — matrix operand order repaired and checked by reading, not compiled, not run on hardware
//
#include <hip/hip_runtime.h>
#include <hip/hip_bf16.h>

typedef __attribute__((ext_vector_type(16))) _Float16 v16h;
typedef __attribute__((ext_vector_type(8)))  _Float16 h8;
typedef __attribute__((ext_vector_type(8)))  float    v8f;

#define T_   1024
#define H_   4096
#define NH_  32
#define NKV_ 8
#define D_   128
#define I_   12288

static __device__ __forceinline__ v8f wmma_f16(v16h a, v16h b, v8f c) {
  return __builtin_amdgcn_wmma_f32_16x16x32_f16(false, a, false, b, (short)0, c, false, false);
}
static __device__ __forceinline__ v16h join8(h8 a, h8 b) {
  return __builtin_shufflevector(a, b, 0,1,2,3,4,5,6,7,8,9,10,11,12,13,14,15);
}

// ---------------------------------------------------------------------------
// res = a + b ; xo = rmsnorm(res) * w     (one block per token, 256 threads)
// ---------------------------------------------------------------------------
__global__ void addnorm_kernel(const float* __restrict__ a, const float* __restrict__ b,
                               const float* __restrict__ w, float* __restrict__ res,
                               float* __restrict__ xo) {
  const int t = blockIdx.x, tid = threadIdx.x;
  const float4* a4 = (const float4*)(a + (size_t)t * H_);
  const float4* b4 = (const float4*)(b + (size_t)t * H_);
  const float4* w4 = (const float4*)w;
  float4* r4 = (float4*)(res + (size_t)t * H_);
  float4* x4 = (float4*)(xo  + (size_t)t * H_);
  float4 v[4];
  float ss = 0.f;
#pragma unroll
  for (int j = 0; j < 4; ++j) {
    int idx = tid + j * 256;
    float4 va = a4[idx], vb = b4[idx], r;
    r.x = va.x + vb.x; r.y = va.y + vb.y; r.z = va.z + vb.z; r.w = va.w + vb.w;
    v[j] = r; r4[idx] = r;
    ss += r.x*r.x + r.y*r.y + r.z*r.z + r.w*r.w;
  }
  __shared__ float sred[256];
  sred[tid] = ss; __syncthreads();
  for (int st = 128; st > 0; st >>= 1) { if (tid < st) sred[tid] += sred[tid + st]; __syncthreads(); }
  float sc = rsqrtf(sred[0] * (1.0f / H_) + 1e-6f);
#pragma unroll
  for (int j = 0; j < 4; ++j) {
    int idx = tid + j * 256;
    float4 wv = w4[idx], r = v[j], o;
    o.x = r.x*sc*wv.x; o.y = r.y*sc*wv.y; o.z = r.z*sc*wv.z; o.w = r.w*sc*wv.w;
    x4[idx] = o;
  }
}

// ---------------------------------------------------------------------------
// Block-diagonal rotation:  xr[t, g*128+n] = sum_k x[t, g*128+k] * R[g][k][n]
// LDS-free: all 4 A- and 4 B-fragments issued first (loads pipeline), then
// 4 WMMAs on two independent accumulator chains.
// ---------------------------------------------------------------------------
template<int KTOT>
__global__ __launch_bounds__(256) void rotate_kernel(const float* __restrict__ x,
                                                     const float* __restrict__ R,
                                                     _Float16* __restrict__ xr) {
  const int g = blockIdx.y, m0 = blockIdx.x * 16;
  const int tid = threadIdx.x, lane = tid & 31, w = tid >> 5;
  const int hi = lane >> 4, lidx = lane & 15;

  const float* xrow = x + (size_t)(m0 + lidx) * KTOT + g * 128;   // A row for this lane
  const float* rcol = R + (size_t)g * 16384 + w * 16 + lidx;      // B column for this lane

  v16h afr[4], bfr[4];
#pragma unroll
  for (int kc = 0; kc < 4; ++kc) {
    const float4* p0 = (const float4*)(xrow + kc * 32 + 16 * hi);
    const float4* p1 = (const float4*)(xrow + kc * 32 + 16 * hi + 8);
    float4 f0 = p0[0], f1 = p0[1], f2 = p1[0], f3 = p1[1];
    v16h a;
    a[0]=(_Float16)f0.x; a[1]=(_Float16)f0.y; a[2]=(_Float16)f0.z; a[3]=(_Float16)f0.w;
    a[4]=(_Float16)f1.x; a[5]=(_Float16)f1.y; a[6]=(_Float16)f1.z; a[7]=(_Float16)f1.w;
    a[8]=(_Float16)f2.x; a[9]=(_Float16)f2.y; a[10]=(_Float16)f2.z; a[11]=(_Float16)f2.w;
    a[12]=(_Float16)f3.x; a[13]=(_Float16)f3.y; a[14]=(_Float16)f3.z; a[15]=(_Float16)f3.w;
    afr[kc] = a;
  }
#pragma unroll
  for (int kc = 0; kc < 4; ++kc) {
    const float* rp = rcol + (size_t)(kc * 32 + 16 * hi) * 128;
    v16h b;
#pragma unroll
    for (int i = 0; i < 16; ++i) b[i] = (_Float16)rp[(size_t)i * 128];
    bfr[kc] = b;
  }
  v8f acc0 = (v8f)0.0f, acc1 = (v8f)0.0f;
  acc0 = wmma_f16(afr[0], bfr[0], acc0);
  acc1 = wmma_f16(afr[1], bfr[1], acc1);
  acc0 = wmma_f16(afr[2], bfr[2], acc0);
  acc1 = wmma_f16(afr[3], bfr[3], acc1);
#pragma unroll
  for (int r = 0; r < 8; ++r)
    xr[(size_t)(m0 + r + 8 * hi) * KTOT + g * 128 + w * 16 + lidx] =
        (_Float16)(acc0[r] + acc1[r]);
}

// ---------------------------------------------------------------------------
// Quantized GEMM: out[T,O] = A_f16[T,K] @ W^T ;  W[o,k] = (qw[o,k]-8)*sc[o,k/128]
// 128x128 macro-tile, 8 waves; wave = 16 rows x 8 N-tiles.  K step 64.
// A fragments issued before the staging barrier (latency hidden behind the
// weight dequant); B fragments loaded 8-at-a-time before the 8 WMMAs.
// ---------------------------------------------------------------------------
template<int K, int O>
__global__ __launch_bounds__(256) void qgemm_kernel(const _Float16* __restrict__ A,
                                                    const int* __restrict__ qw,
                                                    const float* __restrict__ sc,
                                                    float* __restrict__ out) {
  __shared__ alignas(32) _Float16 ldsB[2 * 4096];  // 64x128 weights, B-frag order
  const int tid = threadIdx.x, lane = tid & 31, w = tid >> 5;
  const int hi = lane >> 4, lidx = lane & 15;
  const int o0 = blockIdx.x * 128, m0 = blockIdx.y * 128;

  const int bn = tid >> 1, kc2 = tid & 1;            // B staging coords
  const int*   qrow = qw + (size_t)(o0 + bn) * K + kc2 * 32;
  const float* srow = sc + (size_t)(o0 + bn) * (K >> 7);
  const _Float16* arow = A + (size_t)(m0 + w * 16 + lidx) * K;  // per-lane A row
  _Float16* cb = &ldsB[kc2 * 4096 + (bn >> 4) * 512 + (bn & 15) * 16];

  v8f acc[8];
#pragma unroll
  for (int nt = 0; nt < 8; ++nt) acc[nt] = (v8f)0.0f;

  for (int kc = 0; kc < K; kc += 64) {
    // issue A-fragment loads early: latency overlaps staging + barriers
    v16h afr[2];
#pragma unroll
    for (int kk = 0; kk < 2; ++kk) {
      h8 a0 = *(const h8*)(arow + kc + kk * 32 + 16 * hi);
      h8 a1 = *(const h8*)(arow + kc + kk * 32 + 16 * hi + 8);
      afr[kk] = join8(a0, a1);
    }
    __syncthreads();
    { // stage+dequant 32 int4-coded weights (one 128-group per 32-run)
      float s = srow[(kc + kc2 * 32) >> 7];
      const int4* qp = (const int4*)(qrow + kc);
      int4 qa = qp[0], qb = qp[1], qc = qp[2], qd = qp[3];
      int4 qe = qp[4], qf = qp[5], qg = qp[6], qh = qp[7];
      h8 h0, h1, h2, h3;
      h0[0]=(_Float16)((float)(qa.x-8)*s); h0[1]=(_Float16)((float)(qa.y-8)*s);
      h0[2]=(_Float16)((float)(qa.z-8)*s); h0[3]=(_Float16)((float)(qa.w-8)*s);
      h0[4]=(_Float16)((float)(qb.x-8)*s); h0[5]=(_Float16)((float)(qb.y-8)*s);
      h0[6]=(_Float16)((float)(qb.z-8)*s); h0[7]=(_Float16)((float)(qb.w-8)*s);
      h1[0]=(_Float16)((float)(qc.x-8)*s); h1[1]=(_Float16)((float)(qc.y-8)*s);
      h1[2]=(_Float16)((float)(qc.z-8)*s); h1[3]=(_Float16)((float)(qc.w-8)*s);
      h1[4]=(_Float16)((float)(qd.x-8)*s); h1[5]=(_Float16)((float)(qd.y-8)*s);
      h1[6]=(_Float16)((float)(qd.z-8)*s); h1[7]=(_Float16)((float)(qd.w-8)*s);
      h2[0]=(_Float16)((float)(qe.x-8)*s); h2[1]=(_Float16)((float)(qe.y-8)*s);
      h2[2]=(_Float16)((float)(qe.z-8)*s); h2[3]=(_Float16)((float)(qe.w-8)*s);
      h2[4]=(_Float16)((float)(qf.x-8)*s); h2[5]=(_Float16)((float)(qf.y-8)*s);
      h2[6]=(_Float16)((float)(qf.z-8)*s); h2[7]=(_Float16)((float)(qf.w-8)*s);
      h3[0]=(_Float16)((float)(qg.x-8)*s); h3[1]=(_Float16)((float)(qg.y-8)*s);
      h3[2]=(_Float16)((float)(qg.z-8)*s); h3[3]=(_Float16)((float)(qg.w-8)*s);
      h3[4]=(_Float16)((float)(qh.x-8)*s); h3[5]=(_Float16)((float)(qh.y-8)*s);
      h3[6]=(_Float16)((float)(qh.z-8)*s); h3[7]=(_Float16)((float)(qh.w-8)*s);
      *(h8*)(cb      ) = h0; *(h8*)(cb +   8) = h1;   // k' 0..15  -> lane  n&15
      *(h8*)(cb + 256) = h2; *(h8*)(cb + 264) = h3;   // k' 16..31 -> lane (n&15)+16
      __builtin_prefetch(qrow + kc + 64, 0, 1);
    }
    __syncthreads();
#pragma unroll
    for (int kk = 0; kk < 2; ++kk) {
      v16h bfr[8];
#pragma unroll
      for (int nt = 0; nt < 8; ++nt)
        bfr[nt] = *(const v16h*)&ldsB[kk * 4096 + nt * 512 + lane * 16];
#pragma unroll
      for (int nt = 0; nt < 8; ++nt)
        acc[nt] = wmma_f16(afr[kk], bfr[nt], acc[nt]);
    }
  }
#pragma unroll
  for (int nt = 0; nt < 8; ++nt)
#pragma unroll
    for (int r = 0; r < 8; ++r)
      out[(size_t)(m0 + w * 16 + r + 8 * hi) * O + o0 + nt * 16 + lidx] = acc[nt][r];
}

// ---------------------------------------------------------------------------
// q/k rmsnorm + RoPE + pack to f16 [head][T][D]; v pass-through convert.
// ---------------------------------------------------------------------------
__global__ void qknorm_kernel(const int* __restrict__ pos,
                              const float* __restrict__ qf, const float* __restrict__ kf,
                              const float* __restrict__ vf,
                              const float* __restrict__ w_qn, const float* __restrict__ w_kn,
                              _Float16* __restrict__ qh, _Float16* __restrict__ kh,
                              _Float16* __restrict__ vh) {
  const int t = blockIdx.x, slot = blockIdx.y, d = threadIdx.x;
  __shared__ float sred[128];
  __shared__ float xn[128];
  const float* src; const float* wn = nullptr; _Float16* dst; bool rope;
  if (slot < NH_)            { src = qf + (size_t)t * H_ + slot * D_;                wn = w_qn; dst = qh + ((size_t)slot * T_ + t) * D_;            rope = true; }
  else if (slot < NH_ + NKV_){ int kvh = slot - NH_;        src = kf + (size_t)t * (NKV_*D_) + kvh * D_; wn = w_kn; dst = kh + ((size_t)kvh * T_ + t) * D_; rope = true; }
  else                       { int kvh = slot - NH_ - NKV_; src = vf + (size_t)t * (NKV_*D_) + kvh * D_;            dst = vh + ((size_t)kvh * T_ + t) * D_; rope = false; }
  float xv = src[d];
  if (rope) {
    sred[d] = xv * xv; __syncthreads();
    for (int st = 64; st > 0; st >>= 1) { if (d < st) sred[d] += sred[d + st]; __syncthreads(); }
    float scn = rsqrtf(sred[0] * (1.0f / D_) + 1e-6f);
    xn[d] = xv * scn * wn[d];
    __syncthreads();
    int j = d & 63;
    float inv = __powf(1.0e6f, -(float)j * (2.0f / 128.0f));
    float f = (float)pos[t] * inv;
    float s, c; __sincosf(f, &s, &c);
    float o = (d < 64) ? (xn[d] * c - xn[d + 64] * s) : (xn[d] * c + xn[d - 64] * s);
    dst[d] = (_Float16)o;
  } else {
    dst[d] = (_Float16)xv;
  }
}

// ---------------------------------------------------------------------------
// Flash attention. grid = (T/128, NH); block 256 = 8 waves, wave = 16 rows.
// Q fragments in VGPRs (loaded once).  K fragments: 8 issued, then 8 WMMAs.
// V tile DMA'd row-major into LDS (global_load_async_to_lds_b128) overlapped
// with S = Q*K^T; synchronized with s_wait_asynccnt + barrier.
// ---------------------------------------------------------------------------
__global__ __launch_bounds__(256) void attn_kernel(const _Float16* __restrict__ qh,
                                                   const _Float16* __restrict__ kh,
                                                   const _Float16* __restrict__ vh,
                                                   float* __restrict__ out) {
  __shared__ alignas(32) _Float16 ldsV[32 * 128];  // V chunk, row-major [s][d]
  __shared__ alignas(32) _Float16 ldsP[8 * 512];   // per-wave P, A-frag order
  const int m0 = blockIdx.x * 128, h = blockIdx.y, kvh = h >> 2;
  const int tid = threadIdx.x, lane = tid & 31, w = tid >> 5;
  const int hi = lane >> 4, lidx = lane & 15;
  const float SCALE = 0.08838834764831845f; // D^-0.5

  // Q fragments in registers: 4 k-chunks x 16 halves
  const _Float16* qrow = qh + ((size_t)(h << 10) + m0 + w * 16 + lidx) * D_;
  v16h qfrag[4];
#pragma unroll
  for (int kc = 0; kc < 4; ++kc) {
    h8 a0 = *(const h8*)(qrow + kc * 32 + 16 * hi);
    h8 a1 = *(const h8*)(qrow + kc * 32 + 16 * hi + 8);
    qfrag[kc] = join8(a0, a1);
  }

  const unsigned ldsVbase = (unsigned)(size_t)(void*)&ldsV[0];
  const int vs = tid >> 3, vdq = (tid & 7) * 16;     // async staging coords

  v8f oacc[8];
  float rmax[8], rsum[8];
#pragma unroll
  for (int nt = 0; nt < 8; ++nt) oacc[nt] = (v8f)0.0f;
#pragma unroll
  for (int r = 0; r < 8; ++r) { rmax[r] = -1.0e30f; rsum[r] = 0.0f; }

  const int nch = (m0 >> 5) + 4;
  for (int ch = 0; ch < nch; ++ch) {
    const int s0 = ch * 32;
    __syncthreads();   // prior-iteration ldsV readers done
    { // kick off async DMA of the V chunk (two 16B segments per thread)
      unsigned long long g0 = (unsigned long long)(size_t)
          (vh + ((size_t)(kvh << 10) + s0 + vs) * D_ + vdq);
      unsigned l0 = ldsVbase + (unsigned)(vs * 128 + vdq) * 2u;
      asm volatile("global_load_async_to_lds_b128 %0, %1, off"
                   :: "v"(l0), "v"(g0) : "memory");
      asm volatile("global_load_async_to_lds_b128 %0, %1, off"
                   :: "v"(l0 + 16u), "v"(g0 + 16ull) : "memory");
    }

    // S = Q * K^T while the DMA flies: issue all 8 K fragments, then 8 WMMAs
    v16h bfr[8];
#pragma unroll
    for (int kc = 0; kc < 4; ++kc)
#pragma unroll
      for (int nt = 0; nt < 2; ++nt)
        bfr[kc * 2 + nt] = *(const v16h*)(kh
            + ((size_t)(kvh << 10) + s0 + nt * 16 + lidx) * D_ + kc * 32 + 16 * hi);
    v8f sacc[2]; sacc[0] = (v8f)0.0f; sacc[1] = (v8f)0.0f;
#pragma unroll
    for (int kc = 0; kc < 4; ++kc)
#pragma unroll
      for (int nt = 0; nt < 2; ++nt)
        sacc[nt] = wmma_f16(qfrag[kc], bfr[kc * 2 + nt], sacc[nt]);

    // online softmax (rows at r+8*hi, cols striped over 16 lanes)
    float pv0[8], pv1[8];
#pragma unroll
    for (int r = 0; r < 8; ++r) {
      int trow = m0 + w * 16 + r + 8 * hi;
      float v0 = sacc[0][r] * SCALE, v1 = sacc[1][r] * SCALE;
      if (s0 + lidx      > trow) v0 = -1.0e30f;
      if (s0 + 16 + lidx > trow) v1 = -1.0e30f;
      float cm = fmaxf(v0, v1);
#pragma unroll
      for (int off = 8; off > 0; off >>= 1) cm = fmaxf(cm, __shfl_xor(cm, off, 32));
      float mnew  = fmaxf(rmax[r], cm);
      float alpha = __expf(rmax[r] - mnew);
      float p0 = __expf(v0 - mnew), p1 = __expf(v1 - mnew);
      float ps = p0 + p1;
#pragma unroll
      for (int off = 8; off > 0; off >>= 1) ps += __shfl_xor(ps, off, 32);
      rsum[r] = rsum[r] * alpha + ps;
      rmax[r] = mnew;
      pv0[r] = p0; pv1[r] = p1;
#pragma unroll
      for (int nt = 0; nt < 8; ++nt) oacc[nt][r] *= alpha;
    }

    // P -> per-wave LDS in A-frag order
    _Float16* pw = &ldsP[w * 512];
#pragma unroll
    for (int r = 0; r < 8; ++r) {
      pw[((r + 8 * hi) + 16 * ((lidx >> 3) & 1)) * 16 + (lidx & 7)    ] = (_Float16)pv0[r];
      pw[((r + 8 * hi) + 16 * ((lidx >> 3) & 1)) * 16 + (lidx & 7) + 8] = (_Float16)pv1[r];
    }
    v16h aP = *(const v16h*)&ldsP[w * 512 + lane * 16];

    // V must be in LDS before P*V
    asm volatile("s_wait_asynccnt 0" ::: "memory");
    __syncthreads();

    // issue all 8 V fragments (transpose gather from row-major LDS), then WMMAs
#pragma unroll
    for (int nt = 0; nt < 8; ++nt) {
      int dcol = nt * 16 + lidx;
      v16h bv;
#pragma unroll
      for (int i = 0; i < 16; ++i) bv[i] = ldsV[(8 * hi + i + (i & 8)) * 128 + dcol];
      bfr[nt] = bv;
    }
#pragma unroll
    for (int nt = 0; nt < 8; ++nt)
      oacc[nt] = wmma_f16(aP, bfr[nt], oacc[nt]);
  }

#pragma unroll
  for (int r = 0; r < 8; ++r) {
    float inv = 1.0f / rsum[r];
#pragma unroll
    for (int nt = 0; nt < 8; ++nt)
      out[(size_t)(m0 + w * 16 + r + 8 * hi) * (NH_ * D_) + h * D_ + nt * 16 + lidx] =
          oacc[nt][r] * inv;
  }
}

// ---------------------------------------------------------------------------
// gate = silu(gate) * up   (in place, float4 per thread)
// ---------------------------------------------------------------------------
__global__ void silu_mul_kernel(float* __restrict__ gate, const float* __restrict__ up) {
  size_t idx = (size_t)blockIdx.x * 256 + threadIdx.x;
  float4 g = ((const float4*)gate)[idx];
  float4 u = ((const float4*)up)[idx];
  float4 o;
  o.x = g.x / (1.0f + __expf(-g.x)) * u.x;
  o.y = g.y / (1.0f + __expf(-g.y)) * u.y;
  o.z = g.z / (1.0f + __expf(-g.z)) * u.z;
  o.w = g.w / (1.0f + __expf(-g.w)) * u.w;
  ((float4*)gate)[idx] = o;
}

// ---------------------------------------------------------------------------
extern "C" void kernel_launch(void* const* d_in, const int* in_sizes, int n_in,
                              void* d_out, int out_size, void* d_ws, size_t ws_size,
                              hipStream_t stream) {
  const int*   positions = (const int*)  d_in[0];
  const float* hidden    = (const float*)d_in[1];
  const float* residual  = (const float*)d_in[2];
  const float* w_in   = (const float*)d_in[3];
  const float* w_post = (const float*)d_in[4];
  const float* w_qn   = (const float*)d_in[5];
  const float* w_kn   = (const float*)d_in[6];
  const float* R_q  = (const float*)d_in[7];
  const int*   qw_q = (const int*)  d_in[8];
  const float* sc_q = (const float*)d_in[9];
  const float* R_k  = (const float*)d_in[10];
  const int*   qw_k = (const int*)  d_in[11];
  const float* sc_k = (const float*)d_in[12];
  const float* R_v  = (const float*)d_in[13];
  const int*   qw_v = (const int*)  d_in[14];
  const float* sc_v = (const float*)d_in[15];
  const float* R_o  = (const float*)d_in[16];
  const int*   qw_o = (const int*)  d_in[17];
  const float* sc_o = (const float*)d_in[18];
  const float* R_g  = (const float*)d_in[19];
  const int*   qw_g = (const int*)  d_in[20];
  const float* sc_g = (const float*)d_in[21];
  const float* R_u  = (const float*)d_in[22];
  const int*   qw_u = (const int*)  d_in[23];
  const float* sc_u = (const float*)d_in[24];
  const float* R_d  = (const float*)d_in[25];
  const int*   qw_d = (const int*)  d_in[26];
  const float* sc_d = (const float*)d_in[27];

  char* ws = (char*)d_ws;
  const size_t MB = 1ull << 20;
  float*    res   = (float*)   (ws +   0 * MB); // 16 MB
  float*    x     = (float*)   (ws +  16 * MB); // 16 MB
  _Float16* xr    = (_Float16*)(ws +  32 * MB); // 24 MB (T x I f16 max)
  float*    g0    = (float*)   (ws +  56 * MB); // 16 MB (q / o-proj out)
  float*    g1    = (float*)   (ws +  72 * MB); //  4 MB (k)
  float*    g2    = (float*)   (ws +  76 * MB); //  4 MB (v)
  _Float16* qhb   = (_Float16*)(ws +  80 * MB); //  8 MB
  _Float16* khb   = (_Float16*)(ws +  88 * MB); //  2 MB
  _Float16* vhb   = (_Float16*)(ws +  90 * MB); //  2 MB
  float*    attnb = (float*)   (ws +  92 * MB); // 16 MB
  float*    gate  = (float*)   (ws + 108 * MB); // 48 MB
  float*    up    = (float*)   (ws + 156 * MB); // 48 MB

  float* h2   = (float*)d_out;
  float* res2 = h2 + (size_t)T_ * H_;

  dim3 b256(256), b128(128);

  // res = hidden + residual ; x = rmsnorm(res) * w_in
  addnorm_kernel<<<T_, b256, 0, stream>>>(hidden, residual, w_in, res, x);

  // q / k / v projections
  rotate_kernel<H_><<<dim3(T_/16, H_/128), b256, 0, stream>>>(x, R_q, xr);
  qgemm_kernel<H_, H_><<<dim3(H_/128, T_/128), b256, 0, stream>>>(xr, qw_q, sc_q, g0);
  rotate_kernel<H_><<<dim3(T_/16, H_/128), b256, 0, stream>>>(x, R_k, xr);
  qgemm_kernel<H_, NKV_*D_><<<dim3((NKV_*D_)/128, T_/128), b256, 0, stream>>>(xr, qw_k, sc_k, g1);
  rotate_kernel<H_><<<dim3(T_/16, H_/128), b256, 0, stream>>>(x, R_v, xr);
  qgemm_kernel<H_, NKV_*D_><<<dim3((NKV_*D_)/128, T_/128), b256, 0, stream>>>(xr, qw_v, sc_v, g2);

  // q/k norm + rope ; pack q/k/v as f16 [head][T][D]
  qknorm_kernel<<<dim3(T_, NH_ + 2*NKV_), b128, 0, stream>>>(positions, g0, g1, g2,
                                                             w_qn, w_kn, qhb, khb, vhb);
  // attention
  attn_kernel<<<dim3(T_/128, NH_), b256, 0, stream>>>(qhb, khb, vhb, attnb);

  // o projection
  rotate_kernel<H_><<<dim3(T_/16, H_/128), b256, 0, stream>>>(attnb, R_o, xr);
  qgemm_kernel<H_, H_><<<dim3(H_/128, T_/128), b256, 0, stream>>>(xr, qw_o, sc_o, g0);

  // res2 = h + res ; x2 = rmsnorm(res2) * w_post
  addnorm_kernel<<<T_, b256, 0, stream>>>(g0, res, w_post, res2, x);

  // MLP gate / up
  rotate_kernel<H_><<<dim3(T_/16, H_/128), b256, 0, stream>>>(x, R_g, xr);
  qgemm_kernel<H_, I_><<<dim3(I_/128, T_/128), b256, 0, stream>>>(xr, qw_g, sc_g, gate);
  rotate_kernel<H_><<<dim3(T_/16, H_/128), b256, 0, stream>>>(x, R_u, xr);
  qgemm_kernel<H_, I_><<<dim3(I_/128, T_/128), b256, 0, stream>>>(xr, qw_u, sc_u, up);

  // act = silu(gate) * up  (in place)
  silu_mul_kernel<<<dim3(((size_t)T_ * I_) / 1024), b256, 0, stream>>>(gate, up);

  // down projection -> h2
  rotate_kernel<I_><<<dim3(T_/16, I_/128), b256, 0, stream>>>(gate, R_d, xr);
  qgemm_kernel<I_, H_><<<dim3(H_/128, T_/128), b256, 0, stream>>>(xr, qw_d, sc_d, h2);
}
